// MixtralSparseMoeBlock_35158602285426
// MI455X (gfx1250) — matrix operand order repaired and checked by reading, not compiled, not run on hardware
//
#include <hip/hip_runtime.h>
#include <stdint.h>

typedef __attribute__((ext_vector_type(16))) __bf16 v16bf;
typedef __attribute__((ext_vector_type(8)))  __bf16 v8bf;
typedef __attribute__((ext_vector_type(8)))  float  v8f;
typedef __attribute__((ext_vector_type(4)))  int    v4i;

constexpr int T_ = 2048;
constexpr int H_ = 2048;
constexpr int F_ = 4096;
constexpr int E_ = 8;
constexpr int K_ = 2;
constexpr int P_ = T_ * K_;          // 4096 (token,k) pairs; each pair hits exactly one expert

constexpr int BM  = 128;             // rows (pairs) per block
constexpr int BN1 = 64;              // GEMM1 output cols per block (x2 matrices)
constexpr int BN2 = 128;             // GEMM2 output cols per block
constexpr int BK  = 32;              // K-step (matches wmma 16x16x32)
constexpr int LS  = 40;              // LDS row stride in bf16 elems (80B: 16B-aligned + bank skew)

// CDNA5 async global->LDS copy path (ASYNCcnt), used where no dtype conversion is needed.
// Probe-derived signature: (int4 addrspace(1)*, int4 addrspace(3)*, imm offset, imm cpol)
#if defined(__AMDGCN__) && __has_builtin(__builtin_amdgcn_global_load_async_to_lds_b128)
#define USE_ASYNC_LDS 1
typedef __attribute__((address_space(1))) void* as1vp;
typedef __attribute__((address_space(3))) void* as3vp;
typedef __attribute__((address_space(1))) v4i*  gv4p;
typedef __attribute__((address_space(3))) v4i*  lv4p;
#else
#define USE_ASYNC_LDS 0
#endif

__device__ __forceinline__ void wait_async_lds() {
#if USE_ASYNC_LDS
#if __has_builtin(__builtin_amdgcn_s_wait_asynccnt)
  __builtin_amdgcn_s_wait_asynccnt(0);
#else
  asm volatile("s_wait_asynccnt 0" ::: "memory");
#endif
#endif
}

__device__ __forceinline__ v16bf cat8(v8bf lo, v8bf hi) {
  return __builtin_shufflevector(lo, hi, 0,1,2,3,4,5,6,7,8,9,10,11,12,13,14,15);
}
__device__ __forceinline__ v8f wmma_bf16(v16bf a, v16bf b, v8f c) {
  return __builtin_amdgcn_wmma_f32_16x16x32_bf16(false, a, false, b, (short)0, c, false, false);
}

// ---------------- routing: per-expert ordered pair lists (deterministic) ----------------
__global__ void moe_route(const long long* __restrict__ sel,   // [T][K] int64
                          int* __restrict__ counts,            // [E]
                          int* __restrict__ pairs) {           // [E][P]
  int e = threadIdx.x;
  if (e >= E_) return;
  int c = 0;
  for (int p = 0; p < P_; ++p) {
    if ((int)sel[p] == e) pairs[e * P_ + c++] = p;
  }
  counts[e] = c;
}

// ---------------- GEMM1: act = silu(x@w1[e]) * (x@w2[e]), bf16 WMMA, fp32 accum ----------------
// grid: (m-tiles fastest, n-strips, experts) so concurrent m-blocks share each weight strip in L2
__global__ __launch_bounds__(256) void moe_gemm1(
    const float* __restrict__ x,      // [T][H] fp32
    const float* __restrict__ w1g,    // [E][H][F] fp32
    const float* __restrict__ w2g,    // [E][H][F] fp32
    const int* __restrict__ counts,
    const int* __restrict__ pairs,
    __bf16* __restrict__ act)         // [P][F] bf16
{
  __shared__ __align__(16) __bf16 Al[2][BM][LS];
  __shared__ __align__(16) __bf16 B1[2][BN1][LS];  // transposed: B[n][k]
  __shared__ __align__(16) __bf16 B2[2][BN1][LS];
  __shared__ int rowPair[BM];
  __shared__ int rowTok[BM];

  const int e   = blockIdx.z;
  const int cnt = counts[e];
  const int m0  = blockIdx.x * BM;
  if (m0 >= cnt) return;
  const int n0  = blockIdx.y * BN1;
  const int tid = threadIdx.x;

  if (tid < BM) {
    int idx = m0 + tid;
    if (idx >= cnt) idx = cnt - 1;          // clamp: loads valid, stores guarded below
    int p = pairs[e * P_ + idx];
    rowPair[tid] = p;
    rowTok[tid]  = p >> 1;                  // pair p = t*K + k, K=2
  }
  __syncthreads();

  const float* w1p = w1g + (size_t)e * H_ * F_;
  const float* w2p = w2g + (size_t)e * H_ * F_;

  const int fa_r = tid >> 1, fa_h = tid & 1;   // A fill: 128 rows x 2 halves of 16 floats
  const int fb_k = tid >> 3, fb_n = tid & 7;   // B fill: 32 k x 8 n-octets

  auto fillA = [&](int s, int k0) {
    const float4* src = (const float4*)(x + (size_t)rowTok[fa_r] * H_ + k0 + fa_h * 16);
    float4 f0 = src[0], f1 = src[1], f2 = src[2], f3 = src[3];
    v8bf b0, b1;
    b0[0]=(__bf16)f0.x; b0[1]=(__bf16)f0.y; b0[2]=(__bf16)f0.z; b0[3]=(__bf16)f0.w;
    b0[4]=(__bf16)f1.x; b0[5]=(__bf16)f1.y; b0[6]=(__bf16)f1.z; b0[7]=(__bf16)f1.w;
    b1[0]=(__bf16)f2.x; b1[1]=(__bf16)f2.y; b1[2]=(__bf16)f2.z; b1[3]=(__bf16)f2.w;
    b1[4]=(__bf16)f3.x; b1[5]=(__bf16)f3.y; b1[6]=(__bf16)f3.z; b1[7]=(__bf16)f3.w;
    *(v8bf*)&Al[s][fa_r][fa_h * 16]     = b0;
    *(v8bf*)&Al[s][fa_r][fa_h * 16 + 8] = b1;
  };

  auto fillB = [&](int s, int k0) {
    {
      const float4* src = (const float4*)(w1p + (size_t)(k0 + fb_k) * F_ + n0 + fb_n * 8);
      float4 a = src[0], b = src[1];
      B1[s][fb_n*8+0][fb_k] = (__bf16)a.x; B1[s][fb_n*8+1][fb_k] = (__bf16)a.y;
      B1[s][fb_n*8+2][fb_k] = (__bf16)a.z; B1[s][fb_n*8+3][fb_k] = (__bf16)a.w;
      B1[s][fb_n*8+4][fb_k] = (__bf16)b.x; B1[s][fb_n*8+5][fb_k] = (__bf16)b.y;
      B1[s][fb_n*8+6][fb_k] = (__bf16)b.z; B1[s][fb_n*8+7][fb_k] = (__bf16)b.w;
    }
    {
      const float4* src = (const float4*)(w2p + (size_t)(k0 + fb_k) * F_ + n0 + fb_n * 8);
      float4 a = src[0], b = src[1];
      B2[s][fb_n*8+0][fb_k] = (__bf16)a.x; B2[s][fb_n*8+1][fb_k] = (__bf16)a.y;
      B2[s][fb_n*8+2][fb_k] = (__bf16)a.z; B2[s][fb_n*8+3][fb_k] = (__bf16)a.w;
      B2[s][fb_n*8+4][fb_k] = (__bf16)b.x; B2[s][fb_n*8+5][fb_k] = (__bf16)b.y;
      B2[s][fb_n*8+6][fb_k] = (__bf16)b.z; B2[s][fb_n*8+7][fb_k] = (__bf16)b.w;
    }
  };

  // waves: 4 (rows) x 2 (cols); each wave owns 32 rows x 32 cols (x2 matrices)
  const int wv = tid >> 5, ln = tid & 31, lr = ln & 15, g = ln >> 4;
  const int wm = wv & 3, wn = wv >> 2;

  v8f acc1[2][2] = {};
  v8f acc2[2][2] = {};

  auto compute = [&](int s) {
    const int ra = wm * 32 + lr;
    v16bf af0 = cat8(*(const v8bf*)&Al[s][ra][g*8],      *(const v8bf*)&Al[s][ra][16 + g*8]);
    v16bf af1 = cat8(*(const v8bf*)&Al[s][ra + 16][g*8], *(const v8bf*)&Al[s][ra + 16][16 + g*8]);
    const int cb = wn * 32 + lr;
    v16bf bA0 = cat8(*(const v8bf*)&B1[s][cb][g*8],      *(const v8bf*)&B1[s][cb][16 + g*8]);
    v16bf bA1 = cat8(*(const v8bf*)&B1[s][cb + 16][g*8], *(const v8bf*)&B1[s][cb + 16][16 + g*8]);
    v16bf bB0 = cat8(*(const v8bf*)&B2[s][cb][g*8],      *(const v8bf*)&B2[s][cb][16 + g*8]);
    v16bf bB1 = cat8(*(const v8bf*)&B2[s][cb + 16][g*8], *(const v8bf*)&B2[s][cb + 16][16 + g*8]);
    acc1[0][0] = wmma_bf16(af0, bA0, acc1[0][0]);
    acc1[1][0] = wmma_bf16(af1, bA0, acc1[1][0]);
    acc1[0][1] = wmma_bf16(af0, bA1, acc1[0][1]);
    acc1[1][1] = wmma_bf16(af1, bA1, acc1[1][1]);
    acc2[0][0] = wmma_bf16(af0, bB0, acc2[0][0]);
    acc2[1][0] = wmma_bf16(af1, bB0, acc2[1][0]);
    acc2[0][1] = wmma_bf16(af0, bB1, acc2[0][1]);
    acc2[1][1] = wmma_bf16(af1, bB1, acc2[1][1]);
  };

  fillA(0, 0); fillB(0, 0);
  __syncthreads();
  const int NK = H_ / BK;                       // 64 k-steps
  for (int it = 0; it < NK; ++it) {
    int cur = it & 1;
    if (it + 1 < NK) { fillA(cur ^ 1, (it + 1) * BK); fillB(cur ^ 1, (it + 1) * BK); }
    compute(cur);
    __syncthreads();
  }

  // epilogue: silu(a1)*a2 -> bf16 act rows (guarded against clamped rows)
#pragma unroll
  for (int i = 0; i < 2; ++i) {
#pragma unroll
    for (int j = 0; j < 2; ++j) {
#pragma unroll
      for (int r = 0; r < 8; ++r) {
        int br = wm*32 + i*16 + r + 8*g;        // C layout: M = r + 8*(lane>>4), N = lane&15
        if (m0 + br < cnt) {
          float a1v = acc1[i][j][r], a2v = acc2[i][j][r];
          float hsw = (a1v / (1.0f + __expf(-a1v))) * a2v;
          act[(size_t)rowPair[br] * F_ + n0 + wn*32 + j*16 + lr] = (__bf16)hsw;
        }
      }
    }
  }
}

// ---------------- GEMM2: out2 = act @ w3[e], bf16 WMMA, fp32 out ----------------
__global__ __launch_bounds__(256) void moe_gemm2(
    const __bf16* __restrict__ act,   // [P][F] bf16
    const float* __restrict__ w3g,    // [E][F][H] fp32
    const int* __restrict__ counts,
    const int* __restrict__ pairs,
    float* __restrict__ out2)         // [P][H] fp32
{
  __shared__ __align__(16) __bf16 Al[2][BM][LS];
  __shared__ __align__(16) __bf16 Bl[2][BN2][LS];
  __shared__ int rowPair[BM];

  const int e   = blockIdx.z;
  const int cnt = counts[e];
  const int m0  = blockIdx.x * BM;
  if (m0 >= cnt) return;
  const int n0  = blockIdx.y * BN2;
  const int tid = threadIdx.x;

  if (tid < BM) {
    int idx = m0 + tid;
    if (idx >= cnt) idx = cnt - 1;
    rowPair[tid] = pairs[e * P_ + idx];
  }
  __syncthreads();

  const float* w3p = w3g + (size_t)e * F_ * H_;
  const int fa_r = tid >> 1, fa_h = tid & 1;
  const int fb_k = tid >> 3, fb_n = tid & 7;   // 32 k x 8 n-chunks of 16

  // A is already bf16: copy straight to LDS; async DMA path when available (ASYNCcnt).
  auto fillA = [&](int s, int k0) {
    const __bf16* gsrc = act + (size_t)rowPair[fa_r] * F_ + k0 + fa_h * 16;
    __bf16* ldst = &Al[s][fa_r][fa_h * 16];
#if USE_ASYNC_LDS
    __bf16* gs = const_cast<__bf16*>(gsrc);
    __builtin_amdgcn_global_load_async_to_lds_b128(
        (gv4p)(as1vp)(void*)gs,       (lv4p)(as3vp)(void*)ldst,       0, 0);
    __builtin_amdgcn_global_load_async_to_lds_b128(
        (gv4p)(as1vp)(void*)(gs + 8), (lv4p)(as3vp)(void*)(ldst + 8), 0, 0);
#else
    const v8bf* src = (const v8bf*)gsrc;
    *(v8bf*)ldst       = src[0];
    *(v8bf*)(ldst + 8) = src[1];
#endif
  };
  auto fillB = [&](int s, int k0) {
    const float4* src = (const float4*)(w3p + (size_t)(k0 + fb_k) * H_ + n0 + fb_n * 16);
    float4 a = src[0], b = src[1], c = src[2], d = src[3];
    int nb = fb_n * 16;
    Bl[s][nb+ 0][fb_k] = (__bf16)a.x; Bl[s][nb+ 1][fb_k] = (__bf16)a.y;
    Bl[s][nb+ 2][fb_k] = (__bf16)a.z; Bl[s][nb+ 3][fb_k] = (__bf16)a.w;
    Bl[s][nb+ 4][fb_k] = (__bf16)b.x; Bl[s][nb+ 5][fb_k] = (__bf16)b.y;
    Bl[s][nb+ 6][fb_k] = (__bf16)b.z; Bl[s][nb+ 7][fb_k] = (__bf16)b.w;
    Bl[s][nb+ 8][fb_k] = (__bf16)c.x; Bl[s][nb+ 9][fb_k] = (__bf16)c.y;
    Bl[s][nb+10][fb_k] = (__bf16)c.z; Bl[s][nb+11][fb_k] = (__bf16)c.w;
    Bl[s][nb+12][fb_k] = (__bf16)d.x; Bl[s][nb+13][fb_k] = (__bf16)d.y;
    Bl[s][nb+14][fb_k] = (__bf16)d.z; Bl[s][nb+15][fb_k] = (__bf16)d.w;
  };

  // waves: 4 (rows) x 2 (cols); each wave owns 32 rows x 64 cols
  const int wv = tid >> 5, ln = tid & 31, lr = ln & 15, g = ln >> 4;
  const int wm = wv & 3, wn = wv >> 2;

  v8f acc[2][4] = {};

  auto compute = [&](int s) {
    const int ra = wm * 32 + lr;
    v16bf af0 = cat8(*(const v8bf*)&Al[s][ra][g*8],      *(const v8bf*)&Al[s][ra][16 + g*8]);
    v16bf af1 = cat8(*(const v8bf*)&Al[s][ra + 16][g*8], *(const v8bf*)&Al[s][ra + 16][16 + g*8]);
#pragma unroll
    for (int j = 0; j < 4; ++j) {
      const int cb = wn * 64 + j * 16 + lr;
      v16bf bf = cat8(*(const v8bf*)&Bl[s][cb][g*8], *(const v8bf*)&Bl[s][cb][16 + g*8]);
      acc[0][j] = wmma_bf16(af0, bf, acc[0][j]);
      acc[1][j] = wmma_bf16(af1, bf, acc[1][j]);
    }
  };

  fillA(0, 0); fillB(0, 0);
  wait_async_lds();
  __syncthreads();
  const int NK = F_ / BK;                       // 128 k-steps
  for (int it = 0; it < NK; ++it) {
    int cur = it & 1;
    if (it + 1 < NK) { fillA(cur ^ 1, (it + 1) * BK); fillB(cur ^ 1, (it + 1) * BK); }
    compute(cur);
    wait_async_lds();                           // own async copies done; barrier publishes them
    __syncthreads();
  }

#pragma unroll
  for (int i = 0; i < 2; ++i) {
#pragma unroll
    for (int j = 0; j < 4; ++j) {
#pragma unroll
      for (int r = 0; r < 8; ++r) {
        int br = wm*32 + i*16 + r + 8*g;
        if (m0 + br < cnt)
          out2[(size_t)rowPair[br] * H_ + n0 + wn*64 + j*16 + lr] = acc[i][j][r];
      }
    }
  }
}

// ---------------- reduce: out[t] = rw[t,0]*out2[2t] + rw[t,1]*out2[2t+1] ----------------
__global__ void moe_reduce(const float* __restrict__ rw,
                           const float* __restrict__ out2,
                           float* __restrict__ out) {
  int i = blockIdx.x * blockDim.x + threadIdx.x;     // over T*H/4
  if (i >= T_ * H_ / 4) return;
  int i4 = i * 4;
  int t  = i4 >> 11;                                  // / H_
  int h  = i4 & (H_ - 1);
  const float4 u = *(const float4*)(out2 + (size_t)(2 * t) * H_ + h);
  const float4 v = *(const float4*)(out2 + (size_t)(2 * t) * H_ + H_ + h);
  float g0 = rw[2 * t], g1 = rw[2 * t + 1];
  float4 r;
  r.x = g0 * u.x + g1 * v.x;
  r.y = g0 * u.y + g1 * v.y;
  r.z = g0 * u.z + g1 * v.z;
  r.w = g0 * u.w + g1 * v.w;
  *(float4*)(out + i4) = r;
}

extern "C" void kernel_launch(void* const* d_in, const int* in_sizes, int n_in,
                              void* d_out, int out_size, void* d_ws, size_t ws_size,
                              hipStream_t stream) {
  (void)in_sizes; (void)n_in; (void)out_size; (void)ws_size;

  const float*     x   = (const float*)d_in[0];       // [T][H]
  const long long* sel = (const long long*)d_in[1];   // [T][K] int64
  const float*     rw  = (const float*)d_in[2];       // [T][K]
  const float*     w1  = (const float*)d_in[3];       // [E][H][F]
  const float*     w2  = (const float*)d_in[4];       // [E][H][F]
  const float*     w3  = (const float*)d_in[5];       // [E][F][H]
  float* out = (float*)d_out;                          // [T][H]

  // workspace layout
  char* ws = (char*)d_ws;
  int*    counts = (int*)ws;                                          // 256 B reserved
  int*    pairs  = (int*)(ws + 256);                                  // E*P ints = 128 KB
  size_t  act_off  = 256 + (size_t)E_ * P_ * sizeof(int);             // 131328 (256-aligned)
  __bf16* act    = (__bf16*)(ws + act_off);                           // P*F bf16 = 32 MB
  float*  out2   = (float*)(ws + act_off + (size_t)P_ * F_ * 2);      // P*H fp32 = 32 MB

  moe_route<<<1, 32, 0, stream>>>(sel, counts, pairs);

  dim3 g1(P_ / BM, F_ / BN1, E_);    // m-tiles fastest; most early-exit via counts
  moe_gemm1<<<g1, 256, 0, stream>>>(x, w1, w2, counts, pairs, act);

  dim3 g2(P_ / BM, H_ / BN2, E_);    // 32 x 16 x 8
  moe_gemm2<<<g2, 256, 0, stream>>>(act, w3, counts, pairs, out2);

  int total4 = T_ * H_ / 4;
  moe_reduce<<<(total4 + 255) / 256, 256, 0, stream>>>(rw, out2, out);
}
